// PHNNModel_29171417874825
// MI455X (gfx1250) — matrix operand order repaired and checked by reading, not compiled, not run on hardware
//
#include <hip/hip_runtime.h>

// PHNN fused kernel for MI455X (gfx1250, wave32, WMMA).
// One wave per 16-row batch tile. All GEMMs via v_wmma_f32_16x16x32_f16,
// f32 accumulate, activations staged in LDS as f16.
//  - Global memory through forced addrspace(1) pointers -> global_load_b128/b32
//    (LOADcnt only, decoupled from LDS pipe).
//  - LDS through forced addrspace(3) pointers -> ds_load_b128/ds_store
//    (DScnt only; generic-pointer low 32 bits == LDS offset per ISA 10.2).

typedef __attribute__((ext_vector_type(16))) _Float16 v16h;
typedef __attribute__((ext_vector_type(8)))  _Float16 v8h;
typedef __attribute__((ext_vector_type(4)))  _Float16 v4h;
typedef __attribute__((ext_vector_type(2)))  _Float16 v2h;
typedef __attribute__((ext_vector_type(8)))  float    v8f;
typedef __attribute__((ext_vector_type(4)))  float    v4f;

#define AS1 __attribute__((address_space(1)))
#define AS3 __attribute__((address_space(3)))

typedef AS3 _Float16 lds_h;   // use as lds_h* (32-bit LDS pointers -> ds ops)
typedef AS3 float    lds_f;
typedef AS3 char     lds_c;

union HV { v16h v; _Float16 h[16]; };
union FV { v8f  v; float    f[8];  };

// ---- global-memory accessors (force addrspace(1) -> global_load/global_store)
__device__ __forceinline__ v4f gload4(const float* p) {
  return *(const AS1 v4f*)(unsigned long long)p;
}
__device__ __forceinline__ float gload1(const float* p) {
  return *(const AS1 float*)(unsigned long long)p;
}
__device__ __forceinline__ void gstore1(float* p, float v) {
  *(AS1 float*)(unsigned long long)p = v;
}

__device__ __forceinline__ v8f wmma_f16(v16h a, v16h b, v8f c) {
  // D = A(16x32 f16) x B(32x16 f16) + C(16x16 f32)
  return __builtin_amdgcn_wmma_f32_16x16x32_f16(false, a, false, b, (short)0, c,
                                                false, false);
}

// A fragment (16x32 f16) from LDS row-major [16][lda]: two ds_load_b128 per lane.
// Lane l: m = l&15; half = l>>4 selects K sub-range (contiguous, matches B loaders):
//   half0: K = k0+0..15 ; half1: K = k0+16..31
__device__ __forceinline__ v16h loadA16(const lds_h* buf, int lda, int k0, int lane) {
  const int m  = lane & 15;
  const int kb = (lane >> 4) * 16;
  const lds_h* p = buf + m * lda + k0 + kb;         // 16B aligned by construction
  v8h lo = *(const AS3 v8h*)p;
  v8h hi = *(const AS3 v8h*)(p + 8);
  return __builtin_shufflevector(lo, hi, 0, 1, 2, 3, 4, 5, 6, 7,
                                 8, 9, 10, 11, 12, 13, 14, 15);
}

// B fragment, B[k][n] = W[n0+n][k0+k] (op is X @ W^T), W row-major [N][ldw].
// Contiguous 16 floats per lane -> 4x global_load_b128.
__device__ __forceinline__ v16h loadB_T(const float* W, int ldw, int n0, int k0, int lane) {
  const float* p = W + (size_t)(n0 + (lane & 15)) * ldw + (k0 + (lane >> 4) * 16);
  HV B;
#pragma unroll
  for (int j = 0; j < 4; ++j) {
    v4f q = gload4(p + 4 * j);
#pragma unroll
    for (int e = 0; e < 4; ++e) B.h[4 * j + e] = (_Float16)q[e];
  }
  return B.v;
}

// B fragment, B[k][n] = W[k0+k][n0+n] (op is X @ W), W row-major [K][ldw]. Strided.
__device__ __forceinline__ v16h loadB_N(const float* W, int ldw, int k0, int n0, int lane) {
  const float* p = W + (size_t)(k0 + (lane >> 4) * 16) * ldw + (n0 + (lane & 15));
  HV B;
#pragma unroll
  for (int i = 0; i < 16; ++i) B.h[i] = (_Float16)gload1(p + (size_t)i * ldw);
  return B.v;
}

// ---- Forward: a = in @ W^T + b ; save a (f16) + per-row mu/rsigma ;
// ---- optionally h = silu(LN(a)*g+be) -> hout (f16, ld=ldh). N fixed at 256.
__device__ void dense_ln_silu(const lds_h* inb, int ldin, int K,
                              const float* W, const float* bias,
                              const float* g, const float* be,
                              lds_h* asave, lds_f* mu, lds_f* rs,
                              lds_h* hout, int ldh, int lane) {
  float s1[8], s2[8];
#pragma unroll
  for (int r = 0; r < 8; ++r) { s1[r] = 0.f; s2[r] = 0.f; }
  const int rb = (lane >> 4) * 8;
  const int cl = lane & 15;
#pragma unroll 1
  for (int nt = 0; nt < 16; ++nt) {
    v8f acc = {};
    for (int k0 = 0; k0 < K; k0 += 32)
      acc = wmma_f16(loadA16(inb, ldin, k0, lane), loadB_T(W, K, nt * 16, k0, lane), acc);
    const int col = nt * 16 + cl;
    const float bv = gload1(bias + col);
    FV A; A.v = acc;
#pragma unroll
    for (int r = 0; r < 8; ++r) {
      float av = A.f[r] + bv;                 // accumulator row = rb+r, col = col
      asave[(rb + r) * 256 + col] = (_Float16)av;
      s1[r] += av; s2[r] += av * av;
    }
  }
  __syncthreads();
  // in-lane tile sums done; reduce across the 16-lane group (wave32: masks 1..8)
#pragma unroll
  for (int r = 0; r < 8; ++r) {
#pragma unroll
    for (int msk = 1; msk < 16; msk <<= 1) {
      s1[r] += __shfl_xor(s1[r], msk, 32);
      s2[r] += __shfl_xor(s2[r], msk, 32);
    }
  }
  if (cl == 0) {
#pragma unroll
    for (int r = 0; r < 8; ++r) {
      float m = s1[r] * (1.f / 256.f);
      float v = s2[r] * (1.f / 256.f) - m * m;
      mu[rb + r] = m;
      rs[rb + r] = rsqrtf(v + 1e-5f);
    }
  }
  __syncthreads();
  if (hout) {
    for (int i = lane; i < 16 * 128; i += 32) {
      int m = i >> 7, c = (i & 127) * 2;
      v2h av2 = *(const AS3 v2h*)(asave + m * 256 + c);
      float mu_m = mu[m], rs_m = rs[m];
      v2h o;
#pragma unroll
      for (int e = 0; e < 2; ++e) {
        float av = (float)av2[e];
        float nh = (av - mu_m) * rs_m;
        float l  = nh * gload1(g + c + e) + gload1(be + c + e);
        float sg = 1.f / (1.f + __expf(-l));
        o[e] = (_Float16)(l * sg);
      }
      *(AS3 v2h*)(hout + m * ldh + c) = o;
    }
  }
  __syncthreads();
}

// ---- Backward through one Linear->LN->SiLU layer.
// dh comes either from broadcast Wf[col] (seed) or from da_in @ Wnext (GEMM).
// Produces da (f16) into dbuf ([16][320] used as [16][256]).
__device__ void bwd_ln_silu(const lds_h* da_in, const float* Wnext, const float* Wf,
                            const lds_h* asave, const lds_f* mu, const lds_f* rs,
                            const float* g, const float* be,
                            lds_h* dbuf, lds_f* bm1, lds_f* bm2, int lane) {
  float s1[8], s2[8];
#pragma unroll
  for (int r = 0; r < 8; ++r) { s1[r] = 0.f; s2[r] = 0.f; }
  const int rb = (lane >> 4) * 8;
  const int cl = lane & 15;
#pragma unroll 1
  for (int nt = 0; nt < 16; ++nt) {
    const int col = nt * 16 + cl;
    v8f acc = {};
    if (!Wf) {
      for (int k0 = 0; k0 < 256; k0 += 32)
        acc = wmma_f16(loadA16(da_in, 320, k0, lane),
                       loadB_N(Wnext, 256, k0, nt * 16, lane), acc);
    }
    FV A; A.v = acc;
    const float wfv = Wf ? gload1(Wf + col) : 0.f;
    const float gc  = gload1(g + col);
    const float bec = gload1(be + col);
#pragma unroll
    for (int r = 0; r < 8; ++r) {
      float dh  = Wf ? wfv : A.f[r];
      int  row  = rb + r;
      float av  = (float)asave[row * 256 + col];
      float nh  = (av - mu[row]) * rs[row];
      float l   = nh * gc + bec;
      float sg  = 1.f / (1.f + __expf(-l));
      float sp  = sg * (1.f + l * (1.f - sg));      // silu'(l)
      float dnh = dh * sp * gc;
      dbuf[row * 320 + col] = (_Float16)dnh;
      s1[r] += dnh;
      s2[r] += dnh * nh;
    }
  }
  __syncthreads();
#pragma unroll
  for (int r = 0; r < 8; ++r) {
#pragma unroll
    for (int msk = 1; msk < 16; msk <<= 1) {
      s1[r] += __shfl_xor(s1[r], msk, 32);
      s2[r] += __shfl_xor(s2[r], msk, 32);
    }
  }
  if (cl == 0) {
#pragma unroll
    for (int r = 0; r < 8; ++r) {
      bm1[rb + r] = s1[r] * (1.f / 256.f);
      bm2[rb + r] = s2[r] * (1.f / 256.f);
    }
  }
  __syncthreads();
  // da = rs*(dnh - mean(dnh) - nhat*mean(dnh*nhat)), in-place over dbuf
  for (int i = lane; i < 16 * 128; i += 32) {
    int m = i >> 7, c = (i & 127) * 2;
    v2h dnh2 = *(const AS3 v2h*)(dbuf + m * 320 + c);
    v2h av2  = *(const AS3 v2h*)(asave + m * 256 + c);
    float mu_m = mu[m], rs_m = rs[m], m1 = bm1[m], m2 = bm2[m];
    v2h o;
#pragma unroll
    for (int e = 0; e < 2; ++e) {
      float nh = ((float)av2[e] - mu_m) * rs_m;
      float da = rs_m * ((float)dnh2[e] - m1 - nh * m2);
      o[e] = (_Float16)da;
    }
    *(AS3 v2h*)(dbuf + m * 320 + c) = o;
  }
  __syncthreads();
}

// sigmoid(x @ SW^T + Sb) into out[16][320] f16, cols 256..319 := 1 (the "ones" concat)
__device__ void gate_sigmoid(const lds_h* xt, const float* SW, const float* Sb,
                             lds_h* out, int lane) {
  const int rb = (lane >> 4) * 8;
  const int cl = lane & 15;
#pragma unroll 1
  for (int nt = 0; nt < 16; ++nt) {
    v8f acc = {};
    acc = wmma_f16(loadA16(xt, 64,  0, lane), loadB_T(SW, 64, nt * 16,  0, lane), acc);
    acc = wmma_f16(loadA16(xt, 64, 32, lane), loadB_T(SW, 64, nt * 16, 32, lane), acc);
    const int col = nt * 16 + cl;
    const float bv = gload1(Sb + col);
    FV A; A.v = acc;
#pragma unroll
    for (int r = 0; r < 8; ++r) {
      float v = A.f[r] + bv;
      out[(rb + r) * 320 + col] = (_Float16)(1.f / (1.f + __expf(-v)));
    }
  }
  for (int i = lane; i < 16 * 32; i += 32) {
    int m = i >> 5, c = (i & 31) * 2;
    v2h o; o[0] = (_Float16)1.f; o[1] = (_Float16)1.f;
    *(AS3 v2h*)(out + m * 320 + 256 + c) = o;
  }
  __syncthreads();
}

// out[m][n] = sign * (gH @ op(W))[m][n] * gate[m][n], N=320, K=64.
// mode 0: op = W^T (W is [320][64]);  mode 1: op = W (W is [64][320]).
__device__ void gated_prod(const lds_h* gH, const float* W, int mode, float sign,
                           const lds_h* gate, lds_h* out, int lane) {
  const int rb = (lane >> 4) * 8;
  const int cl = lane & 15;
#pragma unroll 1
  for (int nt = 0; nt < 20; ++nt) {
    v8f acc = {};
    for (int k0 = 0; k0 < 64; k0 += 32) {
      v16h b = mode ? loadB_N(W, 320, k0, nt * 16, lane)
                    : loadB_T(W, 64, nt * 16, k0, lane);
      acc = wmma_f16(loadA16(gH, 64, k0, lane), b, acc);
    }
    const int col = nt * 16 + cl;
    FV A; A.v = acc;
#pragma unroll
    for (int r = 0; r < 8; ++r) {
      float gv = (float)gate[(rb + r) * 320 + col];
      out[(rb + r) * 320 + col] = (_Float16)(sign * A.f[r] * gv);
    }
  }
  __syncthreads();
}

struct PhnnParams {
  const float *x, *u;
  const float *H_W0, *H_b0, *H_g0, *H_be0, *H_Wh, *H_bh, *H_gh, *H_beh, *H_Wf;
  const float *G_W0, *G_b0, *G_g0, *G_be0, *G_Wh, *G_bh, *G_gh, *G_beh, *G_Wf, *G_bf;
  const float *J1, *J2, *Js_W, *Js_b, *Rm, *Rs_W, *Rs_b;
  float *xdot, *y;
};

// Manual LDS layout with lifetime-based overlays (~48.8 KB static LDS ->
// 6 single-wave blocks per WGP):
//   [    0, 2048)  xt   : x tile, f16, live whole kernel
//   [ 2048,10240)  a1   : pre-LN saves (H L1, reused as G scratch)
//   [10240,18432)  a2   : pre-LN saves (H L2) / Gmat;  DEAD by J/R phase
//   [18432,26624)  a3   : pre-LN saves (H L3);         DEAD after bwd seed
//   [26624,36864)  bA   : f16 [16][320] ping
//   [36864,47104)  bB   : f16 [16][320] pong
//   [47104,49152)  gH   : grad_H f16
//   [49152,49920)  stats: mu[3][16], rs[3][16], bm1[16], bm2[16], ut[64]
//   bC  = alias [10240,20480)  (a2 + head of a3; only live in J/R phase)
//   itb = alias [20480,24576)  (tail of a3; live from einsum block to end)
#define SMEM_BYTES 49920

__global__ __launch_bounds__(32) void phnn_fused(PhnnParams P) {
  __shared__ __align__(16) char smem_raw[SMEM_BYTES];
  // Generic LDS pointer's low 32 bits are the LDS byte offset (ISA 10.2) ->
  // truncating cast yields a true addrspace(3) base pointer -> ds_* codegen.
  lds_c* sm = (lds_c*)(unsigned)(unsigned long long)&smem_raw[0];

  lds_h* xt  = (lds_h*)(sm + 0);
  lds_h* a1  = (lds_h*)(sm + 2048);
  lds_h* a2  = (lds_h*)(sm + 10240);
  lds_h* a3  = (lds_h*)(sm + 18432);
  lds_h* bA  = (lds_h*)(sm + 26624);
  lds_h* bB  = (lds_h*)(sm + 36864);
  lds_h* gH  = (lds_h*)(sm + 47104);
  lds_f* mu_s = (lds_f*)(sm + 49152);   // 3 x 16
  lds_f* rs_s = (lds_f*)(sm + 49344);   // 3 x 16
  lds_f* bm1  = (lds_f*)(sm + 49536);
  lds_f* bm2  = (lds_f*)(sm + 49600);
  lds_f* ut   = (lds_f*)(sm + 49664);   // 16 x 4
  lds_h* bC  = (lds_h*)(sm + 10240);    // overlay (a2 + a3 head)
  lds_f* itb = (lds_f*)(sm + 20480);    // overlay (a3 tail)

  const int lane = threadIdx.x;
  const int row0 = blockIdx.x * 16;

  // stage x tile (f32 -> f16, 4-wide) and u tile
  for (int i = lane; i < 256; i += 32) {
    v4f q = gload4(P.x + (size_t)row0 * 64 + i * 4);
    v4h o;
#pragma unroll
    for (int e = 0; e < 4; ++e) o[e] = (_Float16)q[e];
    *(AS3 v4h*)(xt + i * 4) = o;
  }
  for (int i = lane; i < 64; i += 32) ut[i] = gload1(P.u + (size_t)row0 * 4 + i);
  __syncthreads();

  // ---------------- H forward (save a + LN stats per layer) ----------------
  dense_ln_silu(xt, 64, 64,   P.H_W0,         P.H_b0,       P.H_g0,       P.H_be0,
                a1, mu_s + 0,  rs_s + 0,  bA, 320, lane);
  dense_ln_silu(bA, 320, 256, P.H_Wh,         P.H_bh,       P.H_gh,       P.H_beh,
                a2, mu_s + 16, rs_s + 16, bB, 320, lane);
  dense_ln_silu(bB, 320, 256, P.H_Wh + 65536, P.H_bh + 256, P.H_gh + 256, P.H_beh + 256,
                a3, mu_s + 32, rs_s + 32, nullptr, 0, lane);   // h3 never needed

  // ---------------- H backward -> grad_H ----------------
  // seed dh3 = Wf ; da3 -> bA ; dh2 = da3 @ Wh1 ; da2 -> bB ; dh1 = da2 @ Wh0 ; da1 -> bA
  bwd_ln_silu(nullptr, nullptr,        P.H_Wf,  a3, mu_s + 32, rs_s + 32,
              P.H_gh + 256, P.H_beh + 256, bA, bm1, bm2, lane);
  bwd_ln_silu(bA, P.H_Wh + 65536, nullptr, a2, mu_s + 16, rs_s + 16,
              P.H_gh,       P.H_beh,       bB, bm1, bm2, lane);
  bwd_ln_silu(bB, P.H_Wh,         nullptr, a1, mu_s + 0,  rs_s + 0,
              P.H_g0,       P.H_be0,       bA, bm1, bm2, lane);
  {
    const int rb = (lane >> 4) * 8, cl = lane & 15;
#pragma unroll 1
    for (int nt = 0; nt < 4; ++nt) {              // grad_H = da1 @ W0  (K=256, N=64)
      v8f acc = {};
      for (int k0 = 0; k0 < 256; k0 += 32)
        acc = wmma_f16(loadA16(bA, 320, k0, lane),
                       loadB_N(P.H_W0, 64, k0, nt * 16, lane), acc);
      FV A; A.v = acc;
      const int col = nt * 16 + cl;
#pragma unroll
      for (int r = 0; r < 8; ++r) gH[(rb + r) * 64 + col] = (_Float16)A.f[r];
    }
  }
  __syncthreads();

  // ---------------- G forward (a1 reused as per-layer scratch) ----------------
  dense_ln_silu(xt, 64, 64,   P.G_W0,         P.G_b0,       P.G_g0,       P.G_be0,
                a1, mu_s + 0, rs_s + 0, bA, 320, lane);
  dense_ln_silu(bA, 320, 256, P.G_Wh,         P.G_bh,       P.G_gh,       P.G_beh,
                a1, mu_s + 0, rs_s + 0, bB, 320, lane);
  dense_ln_silu(bB, 320, 256, P.G_Wh + 65536, P.G_bh + 256, P.G_gh + 256, P.G_beh + 256,
                a1, mu_s + 0, rs_s + 0, bA, 320, lane);
  {
    const int rb = (lane >> 4) * 8, cl = lane & 15;
#pragma unroll 1
    for (int nt = 0; nt < 16; ++nt) {             // Gmat = h3 @ G_Wf^T + bf -> a2
      v8f acc = {};
      for (int k0 = 0; k0 < 256; k0 += 32)
        acc = wmma_f16(loadA16(bA, 320, k0, lane),
                       loadB_T(P.G_Wf, 256, nt * 16, k0, lane), acc);
      const int col = nt * 16 + cl;
      const float bv = gload1(P.G_bf + col);
      FV A; A.v = acc;
#pragma unroll
      for (int r = 0; r < 8; ++r) a2[(rb + r) * 256 + col] = (_Float16)(A.f[r] + bv);
    }
  }
  __syncthreads();

  // input_term[m][d] = sum_u G[m][d*4+u] * u[m][u] ;  y[m][u] = sum_d G[m][d*4+u]*gH[m][d]
  for (int i = lane; i < 16 * 64; i += 32) {
    int m = i >> 6, d = i & 63;
    float s = 0.f;
#pragma unroll
    for (int uu = 0; uu < 4; ++uu)
      s += (float)a2[m * 256 + d * 4 + uu] * ut[m * 4 + uu];
    itb[i] = s;
  }
  for (int idx = lane; idx < 64; idx += 32) {
    int m = idx >> 2, uu = idx & 3;
    float s = 0.f;
    for (int d = 0; d < 64; ++d)
      s += (float)a2[m * 256 + d * 4 + uu] * (float)gH[m * 64 + d];
    gstore1(P.y + (size_t)(row0 + m) * 4 + uu, s);
  }
  __syncthreads();

  // ---------------- J / R bilinear forms ----------------
  gate_sigmoid(xt, P.Js_W, P.Js_b, bB, lane);      // sigJ -> bB
  gated_prod(gH, P.J1, 0,  1.f, bB, bA, lane);     // tL  =  (gH @ J1^T)*sigJ -> bA
  gated_prod(gH, P.J2, 1, -1.f, bB, bC, lane);     // tRn = -(gH @ J2  )*sigJ -> bC
  gate_sigmoid(xt, P.Rs_W, P.Rs_b, bB, lane);      // sigR -> bB (sigJ consumed)
  gated_prod(gH, P.Rm, 0, -1.f, bB, bB, lane);     // tMn = -(gH @ Rm^T)*sigR -> bB (in place)

  // xdot = tL @ J2^T + tRn @ J1 + tMn @ Rm + input_term  (= Jl - Jr - R + Gu)
  {
    const int rb = (lane >> 4) * 8, cl = lane & 15;
#pragma unroll 1
    for (int nt = 0; nt < 4; ++nt) {
      v8f acc = {};
      for (int k0 = 0; k0 < 320; k0 += 32)
        acc = wmma_f16(loadA16(bA, 320, k0, lane), loadB_T(P.J2, 320, nt * 16, k0, lane), acc);
      for (int k0 = 0; k0 < 320; k0 += 32)
        acc = wmma_f16(loadA16(bC, 320, k0, lane), loadB_N(P.J1, 64, k0, nt * 16, lane), acc);
      for (int k0 = 0; k0 < 320; k0 += 32)
        acc = wmma_f16(loadA16(bB, 320, k0, lane), loadB_N(P.Rm, 64, k0, nt * 16, lane), acc);
      const int col = nt * 16 + cl;
      FV A; A.v = acc;
#pragma unroll
      for (int r = 0; r < 8; ++r) {
        int row = rb + r;
        gstore1(P.xdot + (size_t)(row0 + row) * 64 + col, A.f[r] + itb[row * 64 + col]);
      }
    }
  }
}

extern "C" void kernel_launch(void* const* d_in, const int* in_sizes, int n_in,
                              void* d_out, int out_size, void* d_ws, size_t ws_size,
                              hipStream_t stream) {
  (void)n_in; (void)out_size; (void)d_ws; (void)ws_size;
  PhnnParams P;
  P.x    = (const float*)d_in[0];
  P.u    = (const float*)d_in[1];
  P.H_W0 = (const float*)d_in[2];  P.H_b0  = (const float*)d_in[3];
  P.H_g0 = (const float*)d_in[4];  P.H_be0 = (const float*)d_in[5];
  P.H_Wh = (const float*)d_in[6];  P.H_bh  = (const float*)d_in[7];
  P.H_gh = (const float*)d_in[8];  P.H_beh = (const float*)d_in[9];
  P.H_Wf = (const float*)d_in[10]; /* H_bf (d_in[11]) not needed for grad */
  P.G_W0 = (const float*)d_in[12]; P.G_b0  = (const float*)d_in[13];
  P.G_g0 = (const float*)d_in[14]; P.G_be0 = (const float*)d_in[15];
  P.G_Wh = (const float*)d_in[16]; P.G_bh  = (const float*)d_in[17];
  P.G_gh = (const float*)d_in[18]; P.G_beh = (const float*)d_in[19];
  P.G_Wf = (const float*)d_in[20]; P.G_bf  = (const float*)d_in[21];
  P.J1   = (const float*)d_in[22]; P.J2    = (const float*)d_in[23];
  P.Js_W = (const float*)d_in[24]; P.Js_b  = (const float*)d_in[25];
  P.Rm   = (const float*)d_in[26]; P.Rs_W  = (const float*)d_in[27];
  P.Rs_b = (const float*)d_in[28];

  const int B = in_sizes[0] / 64;
  P.xdot = (float*)d_out;
  P.y    = (float*)d_out + (size_t)B * 64;

  phnn_fused<<<dim3(B / 16), dim3(32), 0, stream>>>(P);
}
